// RegulatoryRNN_403726926554
// MI455X (gfx1250) — matrix operand order repaired and checked by reading, not compiled, not run on hardware
//
#include <hip/hip_runtime.h>
#include <hip/hip_bf16.h>
#include <cstdint>

typedef __attribute__((ext_vector_type(16))) _Float16 v16h;
typedef __attribute__((ext_vector_type(8)))  float    v8f;

#define N_TARGETS 64
#define N_STIMULI 16
#define N_OBSERVED 48
#define HIDDEN 128
#define N_CASES 256
#define N_TIME 64
#define DPAD 96              // 80 padded to 3 K-tiles of 32
#define CASES_PER_BLOCK 64
#define GSTRIDE ((N_TIME + 1) * N_TARGETS)   // 65*64 floats per case in d_out

union FragU { uint4 q[2]; v16h h; };

__device__ __forceinline__ uint32_t pack2h(float a, float b) {
    union { _Float16 h[2]; uint32_t u; } p;
    p.h[0] = (_Float16)a; p.h[1] = (_Float16)b;
    return p.u;
}

// A fragment (16x32 f16): lane holds row M=lane&15; khalf=lane>>4.
// v0..7 <- K = kbase + khalf*16 + 0..15
__device__ __forceinline__ v16h load_a_frag(const _Float16* base, int stride,
                                            int row, int kbase, int khalf) {
    FragU f;
    const _Float16* p = base + row * stride + kbase + khalf * 16;
    f.q[0] = *(const uint4*)p;
    f.q[1] = *(const uint4*)(p + 8);
    return f.h;
}

// B fragment: 256 u32 per fragment, lane l owns u32 [l*8 .. l*8+7]
__device__ __forceinline__ v16h load_b_frag(const uint32_t* frag, int lane) {
    FragU f;
    const uint32_t* p = frag + lane * 8;
    f.q[0] = *(const uint4*)p;
    f.q[1] = *(const uint4*)(p + 4);
    return f.h;
}

// ---------------- init: write g0 into time-slot 0 ----------------
__global__ void rnn_init_g0(const float* __restrict__ obs,
                            const float* __restrict__ hid,
                            float* __restrict__ gbuf) {
    int idx = blockIdx.x * 256 + threadIdx.x;           // 256*64
    if (idx < N_CASES * N_TARGETS) {
        int c = idx >> 6, d = idx & 63;
        float v = (d < N_OBSERVED) ? obs[d] : hid[d - N_OBSERVED];
        gbuf[(size_t)c * GSTRIDE + d] = v;
    }
}

// ---------------- pack W1*mask and W2 into B-fragment layout ----------------
#define NB1 (64 * 3 * 8 * 256)
#define NB2 (64 * 4 * 8 * 256)
__global__ void rnn_pack_weights(const float* __restrict__ W1,
                                 const float* __restrict__ lm,
                                 const float* __restrict__ W2,
                                 uint32_t* __restrict__ B1p,
                                 uint32_t* __restrict__ B2p) {
    int idx = blockIdx.x * 256 + threadIdx.x;
    if (idx < NB1) {
        int r = idx & 255, f = idx >> 8;
        int lane = r >> 3, v = r & 7;
        int nt = f & 7, kt = (f >> 3) % 3, n = f / 24;
        int N = nt * 16 + (lane & 15);
        int K = kt * 32 + (lane >> 4) * 16 + 2 * v;
        float x0 = 0.f, x1 = 0.f;
        if (K < 80)     x0 = W1[(size_t)n * 10240 + N * 80 + K]     * lm[n * 80 + K];
        if (K + 1 < 80) x1 = W1[(size_t)n * 10240 + N * 80 + K + 1] * lm[n * 80 + K + 1];
        B1p[idx] = pack2h(x0, x1);
    } else if (idx < NB1 + NB2) {
        int i2 = idx - NB1;
        int r = i2 & 255, f = i2 >> 8;
        int lane = r >> 3, v = r & 7;
        int nt = f & 7, kt = (f >> 3) & 3, n = f >> 5;
        int N = nt * 16 + (lane & 15);
        int K = kt * 32 + (lane >> 4) * 16 + 2 * v;
        float x0 = W2[(size_t)n * 16384 + N * 128 + K];
        float x1 = W2[(size_t)n * 16384 + N * 128 + K + 1];
        B2p[i2] = pack2h(x0, x1);
    }
}

// ---------------- one RNN time step ----------------
// grid: (4 case tiles, 64 networks), block: 256 threads (8 wave32)
__global__ __launch_bounds__(256)
void rnn_step(const float* __restrict__ stimuli,    // (256,64,16)
              float* __restrict__ gbuf,             // (256,65,64) = d_out
              const uint32_t* __restrict__ B1p,
              const uint32_t* __restrict__ B2p,
              const float* __restrict__ b1,
              const float* __restrict__ b2,
              const float* __restrict__ W3,
              const float* __restrict__ b3,
              int t) {
    __shared__ _Float16 Hs [CASES_PER_BLOCK * DPAD];    // 12 KB
    __shared__ _Float16 H1s[CASES_PER_BLOCK * HIDDEN];  // 16 KB
    __shared__ _Float16 H2s[CASES_PER_BLOCK * HIDDEN];  // 16 KB
    __shared__ float w3s[HIDDEN], b1s[HIDDEN], b2s[HIDDEN];

    const int n     = blockIdx.y;
    const int ctile = blockIdx.x;
    const int tid   = threadIdx.x;
    const int lane  = tid & 31;
    const int wave  = tid >> 5;
    const int khalf = lane >> 4;
    const int nlo   = lane & 15;

    if (tid < HIDDEN) {
        w3s[tid] = W3[n * HIDDEN + tid];
        b1s[tid] = b1[n * HIDDEN + tid];
        b2s[tid] = b2[n * HIDDEN + tid];
    }

    // Fill H = [g(t) | stim(t) | 0-pad], f16, one u32 pair per iteration
    for (int idx = tid; idx < CASES_PER_BLOCK * (DPAD / 2); idx += 256) {
        int c = idx / (DPAD / 2);
        int j = idx - c * (DPAD / 2);
        int d = 2 * j;
        int gc = ctile * CASES_PER_BLOCK + c;
        float x0 = 0.f, x1 = 0.f;
        if (d < 64) {
            const float* gp = gbuf + (size_t)gc * GSTRIDE + (size_t)t * N_TARGETS;
            x0 = gp[d]; x1 = gp[d + 1];
        } else if (d < 80) {
            const float* sp = stimuli + (size_t)gc * N_TIME * N_STIMULI + (size_t)t * N_STIMULI;
            x0 = sp[d - 64]; x1 = sp[d - 63];
        }
        Hs[c * DPAD + d]     = (_Float16)x0;
        Hs[c * DPAD + d + 1] = (_Float16)x1;
    }
    __syncthreads();

    // -------- GEMM1: (64x96)x(96x128), wave owns mt = wave>>1, nt in [ (wave&1)*4, +4 )
    const int mt  = wave >> 1;
    const int ntb = (wave & 1) * 4;
    {
        v16h a[3];
        #pragma unroll
        for (int kt = 0; kt < 3; ++kt)
            a[kt] = load_a_frag(Hs, DPAD, mt * 16 + nlo, kt * 32, khalf);
        const uint32_t* Bb = B1p + (size_t)n * 3 * 8 * 256;
        #pragma unroll
        for (int j = 0; j < 4; ++j) {
            int nt = ntb + j;
            v8f c = {};
            #pragma unroll
            for (int kt = 0; kt < 3; ++kt) {
                v16h b = load_b_frag(Bb + (kt * 8 + nt) * 256, lane);
                c = __builtin_amdgcn_wmma_f32_16x16x32_f16(false, a[kt], false, b,
                                                           (short)0, c, false, false);
            }
            int col   = nt * 16 + nlo;
            float bia = b1s[col];
            int rbase = mt * 16 + khalf * 8;
            #pragma unroll
            for (int r = 0; r < 8; ++r) {
                float v = c[r] + bia;
                v = v > 0.f ? v : 0.f;
                H1s[(rbase + r) * HIDDEN + col] = (_Float16)v;
            }
        }
    }
    __syncthreads();

    // -------- GEMM2: (64x128)x(128x128)
    {
        v16h a[4];
        #pragma unroll
        for (int kt = 0; kt < 4; ++kt)
            a[kt] = load_a_frag(H1s, HIDDEN, mt * 16 + nlo, kt * 32, khalf);
        const uint32_t* Bb = B2p + (size_t)n * 4 * 8 * 256;
        #pragma unroll
        for (int j = 0; j < 4; ++j) {
            int nt = ntb + j;
            v8f c = {};
            #pragma unroll
            for (int kt = 0; kt < 4; ++kt) {
                v16h b = load_b_frag(Bb + (kt * 8 + nt) * 256, lane);
                c = __builtin_amdgcn_wmma_f32_16x16x32_f16(false, a[kt], false, b,
                                                           (short)0, c, false, false);
            }
            int col   = nt * 16 + nlo;
            float bia = b2s[col];
            int rbase = mt * 16 + khalf * 8;
            #pragma unroll
            for (int r = 0; r < 8; ++r) {
                float v = c[r] + bia;
                v = v > 0.f ? v : 0.f;
                H2s[(rbase + r) * HIDDEN + col] = (_Float16)v;
            }
        }
    }
    __syncthreads();

    // -------- output layer: o = sigmoid(h2 . W3 + b3); Euler update
    if (tid < CASES_PER_BLOCK) {
        int c  = tid;
        int gc = ctile * CASES_PER_BLOCK + c;
        float acc = b3[n];
        #pragma unroll 8
        for (int k = 0; k < HIDDEN; ++k)
            acc += (float)H2s[c * HIDDEN + k] * w3s[k];
        float o  = 1.0f / (1.0f + __expf(-acc));
        float gp = gbuf[(size_t)gc * GSTRIDE + (size_t)t * N_TARGETS + n];
        gbuf[(size_t)gc * GSTRIDE + (size_t)(t + 1) * N_TARGETS + n] = 0.2f * o + 0.8f * gp;
    }
}

extern "C" void kernel_launch(void* const* d_in, const int* in_sizes, int n_in,
                              void* d_out, int out_size, void* d_ws, size_t ws_size,
                              hipStream_t stream) {
    const float* stimuli  = (const float*)d_in[0];
    const float* linkmask = (const float*)d_in[1];
    const float* obs_init = (const float*)d_in[2];
    const float* hid_init = (const float*)d_in[3];
    const float* W1       = (const float*)d_in[4];
    const float* b1       = (const float*)d_in[5];
    const float* W2       = (const float*)d_in[6];
    const float* b2       = (const float*)d_in[7];
    const float* W3       = (const float*)d_in[8];
    const float* b3       = (const float*)d_in[9];
    float* gbuf = (float*)d_out;

    uint32_t* B1p = (uint32_t*)d_ws;
    uint32_t* B2p = B1p + NB1;

    // g0 into time-slot 0
    rnn_init_g0<<<(N_CASES * N_TARGETS + 255) / 256, 256, 0, stream>>>(obs_init, hid_init, gbuf);

    // pack weights into WMMA B-fragment layout (mask folded into W1)
    rnn_pack_weights<<<(NB1 + NB2 + 255) / 256, 256, 0, stream>>>(W1, linkmask, W2, B1p, B2p);

    // 64 sequential steps (stream ordering carries the recurrence)
    dim3 grid(N_CASES / CASES_PER_BLOCK, N_TARGETS);
    for (int t = 0; t < N_TIME; ++t) {
        rnn_step<<<grid, 256, 0, stream>>>(stimuli, gbuf, B1p, B2p, b1, b2, W3, b3, t);
    }
}
